// HybridGAT_67181878444405
// MI455X (gfx1250) — matrix operand order repaired and checked by reading, not compiled, not run on hardware
//
#include <hip/hip_runtime.h>
#include <hip/hip_bf16.h>

// ---------------------------------------------------------------------------
// CDNA5 (gfx1250) GAT forward: bf16 WMMA GEMMs + atomic segment softmax/agg.
// ---------------------------------------------------------------------------

typedef __attribute__((ext_vector_type(16))) __bf16        v16bf;
typedef __attribute__((ext_vector_type(8)))  float         v8f;
typedef __attribute__((ext_vector_type(4)))  unsigned int  u32x4;
typedef __bf16 bf16_t;

union FragBF { v16bf v; u32x4 q[2]; };

__device__ __forceinline__ float leaky02(float v) { return v > 0.0f ? v : 0.2f * v; }

__device__ __forceinline__ void atomicMaxF(float* addr, float val) {
  if (val >= 0.0f) atomicMax((int*)addr, __float_as_int(val));
  else             atomicMin((unsigned int*)addr, __float_as_uint(val));
}

// ---- conversion kernels ----------------------------------------------------

// row-major f32 [n,kin] -> bf16 [n,kp] (zero pad K)
__global__ void k_cvt_pad_bf16(const float* __restrict__ src, bf16_t* __restrict__ dst,
                               long n, int kin, int kp) {
  long t = (long)blockIdx.x * blockDim.x + threadIdx.x;
  if (t >= n * kp) return;
  int  k = (int)(t % kp);
  long i = t / kp;
  dst[t] = (k < kin) ? (bf16_t)src[i * (long)kin + k] : (bf16_t)0.0f;
}

// f32 W[kin, nout] -> bf16 Wt[nout, kp] (transpose + zero pad K)
__global__ void k_cvt_wT_bf16(const float* __restrict__ w, bf16_t* __restrict__ wt,
                              int kin, int kp, int nout) {
  long t = (long)blockIdx.x * blockDim.x + threadIdx.x;
  if (t >= (long)nout * kp) return;
  int k  = (int)(t % kp);
  int nn = (int)(t / kp);
  wt[t] = (k < kin) ? (bf16_t)w[(long)k * nout + nn] : (bf16_t)0.0f;
}

// ---- WMMA GEMM: C[M,ldc] = A[M,K](bf16, lda=K) * Bt[ldc rows of K](bf16) ---
// One wave -> one 16x16 f32 tile. 4 waves/block along N.
__global__ __launch_bounds__(128)
void k_gemm_bf16(const bf16_t* __restrict__ A, const bf16_t* __restrict__ Bt,
                 float* __restrict__ C, int K, int ldc) {
  const int lane = threadIdx.x & 31;
  const int wv   = threadIdx.x >> 5;
  const int tm   = blockIdx.x;                 // M tile
  const int tn   = blockIdx.y * 4 + wv;        // N tile
  const int r    = lane & 15;                  // M row (A) / N col (B,C)
  const int hh   = lane >> 4;                  // lane half

  const bf16_t* ap = A  + (size_t)(tm * 16 + r) * K;
  const bf16_t* bp = Bt + (size_t)(tn * 16 + r) * K;

  v8f acc = {};
  for (int k0 = 0; k0 < K; k0 += 32) {
    FragBF a, b;
    // A 16x32 layout: lane half 0 -> K {0..7, 16..23}; half 1 -> K {8..15, 24..31}
    a.q[0] = *(const u32x4*)(ap + k0 + hh * 8);
    a.q[1] = *(const u32x4*)(ap + k0 + 16 + hh * 8);
    // B 32x16 layout: lane half 0 -> K {0..7, 16..23}; half 1 -> K {8..15, 24..31}
    b.q[0] = *(const u32x4*)(bp + k0 + hh * 8);
    b.q[1] = *(const u32x4*)(bp + k0 + 16 + hh * 8);
    acc = __builtin_amdgcn_wmma_f32_16x16x32_bf16(false, a.v, false, b.v,
                                                  (short)0, acc, false, false);
  }
  // C/D layout: VGPR v -> M = v + hh*8, N = r
  float* cp = C + (size_t)(tm * 16 + hh * 8) * ldc + tn * 16 + r;
#pragma unroll
  for (int v = 0; v < 8; ++v) cp[(size_t)v * ldc] = acc[v];
}

// ---- per-node attention logits: a_s/a_d[i*heads+h] = <xp[i,h,:], att[h,:]> -
__global__ void k_att(const float* __restrict__ xp, const float* __restrict__ asrc,
                      const float* __restrict__ adst, float* __restrict__ a_s,
                      float* __restrict__ a_d, long n, int heads, int cdim) {
  long gt = (long)blockIdx.x * blockDim.x + threadIdx.x;
  int  lane = (int)(gt & 31);
  long wv   = gt >> 5;
  if (wv >= n * heads) return;
  int h = (int)(wv % heads);
  const float* xr = xp + wv * cdim;
  float ss = 0.0f, sd = 0.0f;
  for (int c = lane; c < cdim; c += 32) {
    float v = xr[c];
    ss += v * asrc[h * cdim + c];
    sd += v * adst[h * cdim + c];
  }
  for (int o = 16; o > 0; o >>= 1) {
    ss += __shfl_down(ss, o, 32);
    sd += __shfl_down(sd, o, 32);
  }
  if (lane == 0) { a_s[wv] = ss; a_d[wv] = sd; }
}

// ---- segment softmax pieces (self-loop handled analytically) ---------------
__global__ void k_self_max_init(const float* __restrict__ a_s, const float* __restrict__ a_d,
                                float* __restrict__ m, long nh) {
  long t = (long)blockIdx.x * blockDim.x + threadIdx.x;
  if (t >= nh) return;
  m[t] = leaky02(a_s[t] + a_d[t]);
}

__global__ void k_edge_max(const int* __restrict__ ei, const float* __restrict__ a_s,
                           const float* __restrict__ a_d, float* __restrict__ m,
                           int E, int heads) {
  int e = blockIdx.x * blockDim.x + threadIdx.x;
  if (e >= E) return;
  int s = ei[e], d = ei[E + e];
  for (int h = 0; h < heads; ++h) {
    float v = leaky02(a_s[(long)s * heads + h] + a_d[(long)d * heads + h]);
    atomicMaxF(&m[(long)d * heads + h], v);
  }
}

__global__ void k_self_z_init(const float* __restrict__ a_s, const float* __restrict__ a_d,
                              const float* __restrict__ m, float* __restrict__ z, long nh) {
  long t = (long)blockIdx.x * blockDim.x + threadIdx.x;
  if (t >= nh) return;
  z[t] = expf(leaky02(a_s[t] + a_d[t]) - m[t]);
}

__global__ void k_edge_z(const int* __restrict__ ei, const float* __restrict__ a_s,
                         const float* __restrict__ a_d, const float* __restrict__ m,
                         float* __restrict__ z, int E, int heads) {
  int e = blockIdx.x * blockDim.x + threadIdx.x;
  if (e >= E) return;
  int s = ei[e], d = ei[E + e];
  for (int h = 0; h < heads; ++h) {
    long dn = (long)d * heads + h;
    float v = leaky02(a_s[(long)s * heads + h] + a_d[dn]);
    atomicAdd(&z[dn], expf(v - m[dn]));
  }
}

// out = self-loop contribution (also initializes out)
__global__ void k_agg_self(const float* __restrict__ xp, const float* __restrict__ a_s,
                           const float* __restrict__ a_d, const float* __restrict__ m,
                           const float* __restrict__ z, float* __restrict__ out,
                           long n, int heads, int cdim) {
  long t = (long)blockIdx.x * blockDim.x + threadIdx.x;
  if (t >= n * heads * cdim) return;
  long nh = t / cdim;
  float w = expf(leaky02(a_s[nh] + a_d[nh]) - m[nh]) / (z[nh] + 1e-16f);
  out[t] = xp[t] * w;
}

// one wave per (edge, head); 8 coalesced atomicAdds per lane (cdim=256)
__global__ void k_edge_agg(const int* __restrict__ ei, const float* __restrict__ xp,
                           const float* __restrict__ a_s, const float* __restrict__ a_d,
                           const float* __restrict__ m, const float* __restrict__ z,
                           float* __restrict__ out, int E, int heads, int cdim) {
  long gt = (long)blockIdx.x * blockDim.x + threadIdx.x;
  int  lane = (int)(gt & 31);
  long wv   = gt >> 5;
  if (wv >= (long)E * heads) return;
  int e = (int)(wv / heads), h = (int)(wv % heads);
  int s = ei[e], d = ei[E + e];
  long sn = (long)s * heads + h, dn = (long)d * heads + h;
  float w = expf(leaky02(a_s[sn] + a_d[dn]) - m[dn]) / (z[dn] + 1e-16f);
  const float* xs = xp + sn * cdim;
  float*       od = out + dn * cdim;
  for (int c = lane; c < cdim; c += 32) atomicAdd(&od[c], xs[c] * w);
}

__global__ void k_bias_elu(float* __restrict__ h, const float* __restrict__ b,
                           long n, int d) {
  long t = (long)blockIdx.x * blockDim.x + threadIdx.x;
  if (t >= n * d) return;
  float v = h[t] + b[t % d];
  h[t] = v > 0.0f ? v : (expf(v) - 1.0f);
}

__global__ void k_zero(float* __restrict__ p, int n) {
  int t = blockIdx.x * blockDim.x + threadIdx.x;
  if (t < n) p[t] = 0.0f;
}

__global__ void k_pool(const float* __restrict__ h, float* __restrict__ pooled,
                       long n, int cdim) {
  int c = threadIdx.x;  // blockDim == cdim
  float s = 0.0f;
  for (long i = blockIdx.x; i < n; i += gridDim.x) s += h[i * cdim + c];
  atomicAdd(&pooled[c], s);
}

__global__ void k_fc(const float* __restrict__ pooled, const float* __restrict__ fcW,
                     const float* __restrict__ fcb, float* __restrict__ out,
                     int cdim, int ncls, float invn) {
  int j = threadIdx.x;
  if (j >= ncls) return;
  float s = fcb[j];
  for (int c = 0; c < cdim; ++c) s += pooled[c] * invn * fcW[c * ncls + j];
  out[j] = s;
}

// ---------------------------------------------------------------------------

extern "C" void kernel_launch(void* const* d_in, const int* in_sizes, int n_in,
                              void* d_out, int out_size, void* d_ws, size_t ws_size,
                              hipStream_t stream) {
  (void)in_sizes; (void)n_in; (void)out_size; (void)ws_size;

  const long N   = 50000;
  const int  E   = 400000;
  const int  INF = 133, KP1 = 160, D1 = 512, H1 = 2, C = 256, K2 = 512, D2 = 256;

  const float* x      = (const float*)d_in[0];
  const int*   ei     = (const int*)  d_in[1];
  const float* W1     = (const float*)d_in[2];
  const float* asrc1  = (const float*)d_in[3];
  const float* adst1  = (const float*)d_in[4];
  const float* b1     = (const float*)d_in[5];
  const float* W2     = (const float*)d_in[6];
  const float* asrc2  = (const float*)d_in[7];
  const float* adst2  = (const float*)d_in[8];
  const float* b2     = (const float*)d_in[9];
  const float* fcW    = (const float*)d_in[10];
  const float* fcb    = (const float*)d_in[11];
  float*       out    = (float*)d_out;

  char* base = (char*)d_ws;
  size_t off = 0;
  auto take = [&](size_t bytes) { size_t o = off; off = (off + bytes + 255) & ~(size_t)255; return o; };

  size_t oXp1  = take((size_t)N * D1 * 4);   // 102.4 MB  (later: h1bf | xp2)
  size_t oOut1 = take((size_t)N * D1 * 4);   // 102.4 MB  (also: xbf early, out2/h2 late)
  size_t oW1t  = take((size_t)KP1 * D1 * 2);
  size_t oW2t  = take((size_t)K2 * D2 * 2);
  size_t oAs1  = take((size_t)N * H1 * 4);
  size_t oAd1  = take((size_t)N * H1 * 4);
  size_t oM1   = take((size_t)N * H1 * 4);
  size_t oZ1   = take((size_t)N * H1 * 4);
  size_t oAs2  = take((size_t)N * 4);
  size_t oAd2  = take((size_t)N * 4);
  size_t oM2   = take((size_t)N * 4);
  size_t oZ2   = take((size_t)N * 4);
  size_t oPool = take(256 * 4);

  float*  xp1  = (float*) (base + oXp1);
  float*  out1 = (float*) (base + oOut1);
  bf16_t* xbf  = (bf16_t*)(base + oOut1);                       // dead before out1 written
  bf16_t* w1t  = (bf16_t*)(base + oW1t);
  bf16_t* w2t  = (bf16_t*)(base + oW2t);
  bf16_t* h1bf = (bf16_t*)(base + oXp1);                        // reuses xp1 region
  float*  xp2  = (float*) (base + oXp1 + (size_t)N * D1 * 2);   // second half of xp1 region
  float*  out2 = (float*) (base + oOut1);                       // reuses out1/h1 region
  float*  as1  = (float*)(base + oAs1); float* ad1 = (float*)(base + oAd1);
  float*  m1   = (float*)(base + oM1);  float* z1  = (float*)(base + oZ1);
  float*  as2  = (float*)(base + oAs2); float* ad2 = (float*)(base + oAd2);
  float*  m2   = (float*)(base + oM2);  float* z2  = (float*)(base + oZ2);
  float*  pooled = (float*)(base + oPool);

  const int B = 256;
  auto blocks = [](long work, int blk) { return (unsigned)((work + blk - 1) / blk); };

  // ---- layer 1 -------------------------------------------------------------
  k_cvt_pad_bf16<<<blocks(N * KP1, B), B, 0, stream>>>(x, xbf, N, INF, KP1);
  k_cvt_wT_bf16 <<<blocks((long)D1 * KP1, B), B, 0, stream>>>(W1, w1t, INF, KP1, D1);
  k_cvt_wT_bf16 <<<blocks((long)D2 * K2, B), B, 0, stream>>>(W2, w2t, K2, K2, D2);

  k_gemm_bf16<<<dim3(3125, D1 / 16 / 4), 128, 0, stream>>>(xbf, w1t, xp1, KP1, D1);

  k_att<<<blocks(N * H1 * 32, B), B, 0, stream>>>(xp1, asrc1, adst1, as1, ad1, N, H1, C);
  k_self_max_init<<<blocks(N * H1, B), B, 0, stream>>>(as1, ad1, m1, N * H1);
  k_edge_max<<<blocks(E, B), B, 0, stream>>>(ei, as1, ad1, m1, E, H1);
  k_self_z_init<<<blocks(N * H1, B), B, 0, stream>>>(as1, ad1, m1, z1, N * H1);
  k_edge_z<<<blocks(E, B), B, 0, stream>>>(ei, as1, ad1, m1, z1, E, H1);
  k_agg_self<<<blocks(N * H1 * C, B), B, 0, stream>>>(xp1, as1, ad1, m1, z1, out1, N, H1, C);
  k_edge_agg<<<blocks((long)E * H1 * 32, B), B, 0, stream>>>(ei, xp1, as1, ad1, m1, z1, out1, E, H1, C);
  k_bias_elu<<<blocks(N * D1, B), B, 0, stream>>>(out1, b1, N, D1);

  // ---- layer 2 -------------------------------------------------------------
  k_cvt_pad_bf16<<<blocks(N * K2, B), B, 0, stream>>>(out1, h1bf, N, K2, K2);
  k_gemm_bf16<<<dim3(3125, D2 / 16 / 4), 128, 0, stream>>>(h1bf, w2t, xp2, K2, D2);

  k_att<<<blocks(N * 32, B), B, 0, stream>>>(xp2, asrc2, adst2, as2, ad2, N, 1, C);
  k_self_max_init<<<blocks(N, B), B, 0, stream>>>(as2, ad2, m2, N);
  k_edge_max<<<blocks(E, B), B, 0, stream>>>(ei, as2, ad2, m2, E, 1);
  k_self_z_init<<<blocks(N, B), B, 0, stream>>>(as2, ad2, m2, z2, N);
  k_edge_z<<<blocks(E, B), B, 0, stream>>>(ei, as2, ad2, m2, z2, E, 1);
  k_agg_self<<<blocks(N * C, B), B, 0, stream>>>(xp2, as2, ad2, m2, z2, out2, N, 1, C);
  k_edge_agg<<<blocks((long)E * 32, B), B, 0, stream>>>(ei, xp2, as2, ad2, m2, z2, out2, E, 1, C);
  k_bias_elu<<<blocks(N * D2, B), B, 0, stream>>>(out2, b2, N, D2);

  // ---- pool + fc -----------------------------------------------------------
  k_zero<<<1, 256, 0, stream>>>(pooled, 256);
  k_pool<<<128, 256, 0, stream>>>(out2, pooled, N, C);
  k_fc<<<1, 32, 0, stream>>>(pooled, fcW, fcb, out, C, 3, 1.0f / (float)N);
}
